// DarcyFlowS4Model_73134703116338
// MI455X (gfx1250) — matrix operand order repaired and checked by reading, not compiled, not run on hardware
//
#include <hip/hip_runtime.h>
#include <hip/hip_bf16.h>
#include <math.h>

// ---------------------------------------------------------------------------
// S4ND Darcy-flow forward for MI455X (gfx1250, wave32, WMMA + TDM).
// Separable Toeplitz-matmul formulation of the 2-D SSM convolution:
//   conv2d(u, kh ⊗ kw)  ==  T_w-conv( T_h-conv(u) ),  T = lower-tri Toeplitz.
// Each 128x128 plane conv is a 128x128x128 GEMM done with
// v_wmma_f32_16x16x32_f16 tiles; GLU projection is a (65536x256)x(256x512)
// WMMA GEMM. Bulk global->LDS tile staging is issued through the Tensor Data
// Mover (tensor_load_to_lds + s_wait_tensorcnt) to keep VMEM/VALU slots free
// for the WMMA stream.
// ---------------------------------------------------------------------------

typedef __attribute__((ext_vector_type(16))) _Float16 v16h;
typedef __attribute__((ext_vector_type(8)))  float    v8f;
typedef __attribute__((ext_vector_type(2)))  _Float16 h2;
typedef unsigned int u32x4 __attribute__((ext_vector_type(4)));
typedef int          i32x4 __attribute__((ext_vector_type(4)));
typedef int          i32x8 __attribute__((ext_vector_type(8)));

#define BATCH   4
#define HDIM    128
#define WDIM    128
#define DMODEL  256
#define NSTATE  64
#define NLAYERS 4
#define HW      (HDIM * WDIM)      // 16384
#define NPIX    (BATCH * HW)       // 65536
#define NPLANES (BATCH * DMODEL)   // 1024

// ---------------------------------------------------------------------------
// Tensor Data Mover: stage a 2-D tile (16-bit elements) global -> LDS.
// D# packed per CDNA5 ISA 8.3/8.4: group0 = {count=1, lds_addr, global_addr,
// type=2}; group1 = {data_size=2B, tensor dims, tile dims, dim0 stride}.
// Higher-dim groups (2/3) are zero (<=2-D tile). 6-arg builtin form on this
// toolchain. Issue once per wave (EXEC-independent); caller waits TENSORcnt.
// ---------------------------------------------------------------------------
__device__ __forceinline__ void tdm_load_2d_f16(unsigned lds_off,
                                                unsigned long long gaddr,
                                                unsigned tw, unsigned th,
                                                unsigned stride_elems) {
  u32x4 g0;
  g0[0] = 1u;                                           // count=1, user mode
  g0[1] = lds_off;                                      // LDS byte address
  g0[2] = (unsigned)gaddr;                              // global_addr[31:0]
  g0[3] = (unsigned)((gaddr >> 32) & 0x01ffffffu) | (2u << 30);  // [56:32]|type=2
  i32x8 g1;
  g1[0] = 1 << 16;                                      // data_size=1 (2 bytes)
  g1[1] = (int)((tw & 0xffffu) << 16);                  // tensor_dim0[15:0]
  g1[2] = (int)(((tw >> 16) & 0xffffu) |                // tensor_dim0[31:16]
                ((th & 0xffffu) << 16));                // tensor_dim1[15:0]
  g1[3] = (int)(((th >> 16) & 0xffffu) |                // tensor_dim1[31:16]
                ((tw & 0xffffu) << 16));                // tile_dim0
  g1[4] = (int)(th & 0xffffu);                          // tile_dim1 (tile_dim2=0)
  g1[5] = (int)stride_elems;                            // tensor_dim0_stride[31:0]
  g1[6] = 0;
  g1[7] = 0;
  i32x4 z4 = {0, 0, 0, 0};
  i32x8 z8 = {0, 0, 0, 0, 0, 0, 0, 0};
  __builtin_amdgcn_tensor_load_to_lds(g0, g1, z4, z4, z8, 0);
}

// ---------------------------------------------------------------------------
// Encoder: h[pix, c] = x[pix]*W_enc[0,c] + grid[pix]*W_enc[1,c] + b_enc[c]
// ---------------------------------------------------------------------------
__global__ __launch_bounds__(256) void encoder_kernel(
    const float* __restrict__ x, const float* __restrict__ grid,
    const float* __restrict__ W_enc, const float* __restrict__ b_enc,
    float* __restrict__ h, _Float16* __restrict__ h16) {
  int idx = blockIdx.x * 256 + threadIdx.x;      // NPIX*DMODEL threads
  int pix = idx >> 8;
  int c   = idx & (DMODEL - 1);
  float v = x[pix] * W_enc[c] + grid[pix] * W_enc[DMODEL + c] + b_enc[c];
  h[idx]   = v;
  h16[idx] = (_Float16)v;
}

// ---------------------------------------------------------------------------
// S4D kernel synthesis for one layer: kf[axis, c, l], axis in {0(H),1(W)}
//   K[c,l] = 2*Re( sum_n C*(exp(dtA)-1)/A * exp(dtA*l) )
// ---------------------------------------------------------------------------
__global__ __launch_bounds__(256) void ssm_kernel(
    const float* __restrict__ log_dt,   // (2, d)
    const float* __restrict__ logA_re,  // (2, d, N)
    const float* __restrict__ A_im,
    const float* __restrict__ C_re,
    const float* __restrict__ C_im,
    float* __restrict__ kf) {           // (2, d, 128)
  int gid = blockIdx.x * 256 + threadIdx.x;   // 2*256*128 threads
  int a   = gid >> 15;
  int rem = gid & 32767;
  int c   = rem >> 7;
  int l   = rem & 127;
  int dci = a * DMODEL + c;
  float dt = expf(log_dt[dci]);
  float fl = (float)l;
  float K  = 0.0f;
  for (int n = 0; n < NSTATE; ++n) {
    int idx   = dci * NSTATE + n;
    float Are = -expf(logA_re[idx]);
    float Aim = A_im[idx];
    float dr  = dt * Are, di = dt * Aim;
    float er  = expf(dr);
    float Ecr = er * cosf(di) - 1.0f;       // Re(exp(dtA) - 1)
    float Eci = er * sinf(di);              // Im(exp(dtA) - 1)
    float den = Are * Are + Aim * Aim;
    float Fr  = (Ecr * Are + Eci * Aim) / den;   // (E-1)/A
    float Fi  = (Eci * Are - Ecr * Aim) / den;
    float cr  = C_re[idx], ci = C_im[idx];
    float CBr = cr * Fr - ci * Fi;
    float CBi = cr * Fi + ci * Fr;
    float pr  = expf(dr * fl);
    float ang = di * fl;
    float Pr  = pr * cosf(ang), Pi = pr * sinf(ang);
    K += CBr * Pr - CBi * Pi;
  }
  kf[dci * HDIM + l] = 2.0f * K;
}

// ---------------------------------------------------------------------------
// (B,HW,d) f16 -> per-channel planes (B*d, HW) f16 via LDS tile transpose.
// ---------------------------------------------------------------------------
__global__ void transpose_hd_kernel(const _Float16* __restrict__ src,
                                    _Float16* __restrict__ dst) {
  __shared__ _Float16 tile[32][33];
  int b  = blockIdx.z;
  int p0 = blockIdx.x * 32;
  int c0 = blockIdx.y * 32;
  for (int i = threadIdx.y; i < 32; i += 8)
    tile[i][threadIdx.x] = src[((size_t)b * HW + p0 + i) * DMODEL + c0 + threadIdx.x];
  __syncthreads();
  for (int i = threadIdx.y; i < 32; i += 8)
    dst[((size_t)(b * DMODEL + c0 + i)) * HW + p0 + threadIdx.x] =
        tile[threadIdx.x][i];
}

// ---------------------------------------------------------------------------
// Toeplitz conv pass over one 128x128 plane per workgroup (8 waves).
//   OUT = T(k[c]) @ IN,  T lower-triangular Toeplitz, via WMMA f16.
// Plane is staged global->LDS by the TDM (32KB contiguous), then transposed
// in-LDS for the WMMA B-operand layout. A-tile (16x32 of T) is synthesized
// per-lane from the 128-entry kernel vector (causality folded in as zeros;
// waves skip K-tiles entirely above their row strip).
// pass 0 (H-axis): write OUT transposed  -> y1t[j][i]
// pass 1 (W-axis): OUT is already Y2^T; fuse  + Dskip*u, GELU, scatter to
//                  pixel-major g16 for the GLU GEMM.
// ---------------------------------------------------------------------------
__global__ __launch_bounds__(256) void conv_pass_kernel(
    const _Float16* __restrict__ in_planes,
    const float* __restrict__ kf_axis,      // (d, 128)
    _Float16* __restrict__ outT,            // pass 0 output planes
    const _Float16* __restrict__ u_planes,  // pass 1: skip input planes
    const float* __restrict__ Dskip,        // pass 1: (d)
    _Float16* __restrict__ g16,             // pass 1: (B,HW,d)
    int pass) {
  __shared__ _Float16 Ut[HDIM * WDIM];    // transposed plane (B operand)
  __shared__ _Float16 Uraw[HDIM * WDIM];  // TDM landing buffer (reused for kb)
  int bc  = blockIdx.x;
  int c   = bc & (DMODEL - 1);
  int b   = bc >> 8;
  int tid = threadIdx.x;
  const _Float16* IN = in_planes + (size_t)bc * HW;

  // --- TDM: pull the whole 128x128 f16 plane into LDS (one wave issues) ---
  if (tid < 32) {
    tdm_load_2d_f16((unsigned)(uintptr_t)Uraw,
                    (unsigned long long)(uintptr_t)IN, HW, 1, HW);
    __builtin_amdgcn_s_wait_tensorcnt((short)0);
  }
  __syncthreads();
  // in-LDS transpose: Ut[n][k] = IN[k][n]
  for (int idx = tid; idx < HW; idx += 256) {
    int p = idx >> 7, j = idx & 127;
    Ut[j * WDIM + p] = Uraw[idx];
  }
  __syncthreads();
  float* kb = (float*)Uraw;               // Uraw is dead; reuse for kernel vec
  for (int t = tid; t < HDIM; t += 256) kb[t] = kf_axis[c * HDIM + t];
  __syncthreads();

  int wave = tid >> 5, lane = tid & 31;
  int ln = lane & 15, hi = lane >> 4;
  int m0  = wave * 16;
  int row = m0 + ln;
  v8f acc[8] = {};
  for (int k0 = 0; k0 < HDIM; k0 += 32) {
    if (k0 > m0 + 15) break;            // causality: T strip is zero beyond row
    // synthesize A tile: A[m][k] = (row>=k) ? k_vec[row-k] : 0
    v16h Aa;
#pragma unroll
    for (int v = 0; v < 8; ++v) {
      int kk = k0 + 2 * v + 8 * (v >> 2) + 8 * hi;   // f16 A VGPR layout
      Aa[2 * v]     = (row >= kk)     ? (_Float16)kb[row - kk]     : (_Float16)0.0f;
      Aa[2 * v + 1] = (row >= kk + 1) ? (_Float16)kb[row - kk - 1] : (_Float16)0.0f;
    }
#pragma unroll
    for (int nt = 0; nt < 8; ++nt) {
      const _Float16* brow = &Ut[(nt * 16 + ln) * WDIM + k0 + 8 * hi];
      v16h Bb;
#pragma unroll
      for (int v = 0; v < 8; ++v) {
        h2 p = *(const h2*)(brow + 2 * v + 8 * (v >> 2));
        Bb[2 * v] = p[0]; Bb[2 * v + 1] = p[1];
      }
      acc[nt] = __builtin_amdgcn_wmma_f32_16x16x32_f16(
          false, Aa, false, Bb, (short)0, acc[nt], false, false);
    }
  }

  if (pass == 0) {
    _Float16* OUTP = outT + (size_t)bc * HW;   // y1t[j][i]
#pragma unroll
    for (int nt = 0; nt < 8; ++nt) {
      int j = nt * 16 + ln;
#pragma unroll
      for (int r = 0; r < 8; ++r) {
        int i = m0 + r + 8 * hi;
        OUTP[j * HDIM + i] = (_Float16)acc[nt][r];
      }
    }
  } else {
    const _Float16* UP = u_planes + (size_t)bc * HW;
    float ds = Dskip[c];
#pragma unroll
    for (int nt = 0; nt < 8; ++nt) {
      int i = nt * 16 + ln;               // C cols = H index
#pragma unroll
      for (int r = 0; r < 8; ++r) {
        int j = m0 + r + 8 * hi;          // C rows = W index
        float val = acc[nt][r] + ds * (float)UP[i * WDIM + j];
        float t  = 0.7978845608f * (val + 0.044715f * val * val * val);
        float gl = 0.5f * val * (1.0f + tanhf(t));   // tanh GELU
        g16[((size_t)b * HW + i * WDIM + j) * DMODEL + c] = (_Float16)gl;
      }
    }
  }
}

// ---------------------------------------------------------------------------
// GLU GEMM: Y = g16 (M x 256) @ W_out (256 x 512); each WG owns a 128-row x
// 64-channel tile and computes BOTH halves (a = cols c, g = cols 256+c) so the
// sigmoid gate pairing is register-local. The 128x32 activation tile is a
// strided 2-D tile staged by the TDM each K step; weight tiles are staged by
// threads (f32 -> f16 conversion). Fused bias + GLU + residual add.
// ---------------------------------------------------------------------------
__global__ __launch_bounds__(256) void glu_gemm_kernel(
    const _Float16* __restrict__ g16,
    const float* __restrict__ Wout,   // (256, 512), layer slice
    const float* __restrict__ bout,   // (512)
    const float* __restrict__ hres,   // (M, 256) f32 residual
    float* __restrict__ rbuf) {       // (M, 256) pre-norm output
  __shared__ _Float16 At[128 * 32];
  __shared__ _Float16 Ba[64 * 32];
  __shared__ _Float16 Bg[64 * 32];
  int row0 = blockIdx.x * 128;
  int c0   = blockIdx.y * 64;
  int tid  = threadIdx.x;
  int wave = tid >> 5, lane = tid & 31;
  int ln = lane & 15, hi = lane >> 4;
  int m0 = wave * 16;
  v8f accA[4] = {};
  v8f accG[4] = {};
  for (int kk0 = 0; kk0 < DMODEL; kk0 += 32) {
    // TDM: stage the 128x32 f16 activation tile (row stride 256 elems).
    if (tid < 32) {
      tdm_load_2d_f16((unsigned)(uintptr_t)At,
                      (unsigned long long)(uintptr_t)(g16 + (size_t)row0 * DMODEL + kk0),
                      32, 128, DMODEL);
      __builtin_amdgcn_s_wait_tensorcnt((short)0);
    }
    if (kk0 + 32 < DMODEL)   // pull next weight rows toward the WGP early
      __builtin_prefetch(&Wout[(size_t)(kk0 + 32) * (2 * DMODEL) + c0], 0, 1);
    for (int idx = tid; idx < 64 * 32; idx += 256) {
      int cc = idx >> 5, k = idx & 31;
      const float* wrow = &Wout[(size_t)(kk0 + k) * (2 * DMODEL) + c0 + cc];
      Ba[idx] = (_Float16)wrow[0];
      Bg[idx] = (_Float16)wrow[DMODEL];
    }
    __syncthreads();
    v16h Aa;
#pragma unroll
    for (int v = 0; v < 8; ++v) {
      int kk = 2 * v + 8 * (v >> 2) + 8 * hi;
      h2 p = *(const h2*)(&At[(m0 + ln) * 32 + kk]);
      Aa[2 * v] = p[0]; Aa[2 * v + 1] = p[1];
    }
#pragma unroll
    for (int ct = 0; ct < 4; ++ct) {
      v16h Bta, Btg;
#pragma unroll
      for (int v = 0; v < 8; ++v) {
        int kk = 2 * v + 8 * (v >> 2) + 8 * hi;
        h2 pa = *(const h2*)(&Ba[(ct * 16 + ln) * 32 + kk]);
        h2 pg = *(const h2*)(&Bg[(ct * 16 + ln) * 32 + kk]);
        Bta[2 * v] = pa[0]; Bta[2 * v + 1] = pa[1];
        Btg[2 * v] = pg[0]; Btg[2 * v + 1] = pg[1];
      }
      accA[ct] = __builtin_amdgcn_wmma_f32_16x16x32_f16(
          false, Aa, false, Bta, (short)0, accA[ct], false, false);
      accG[ct] = __builtin_amdgcn_wmma_f32_16x16x32_f16(
          false, Aa, false, Btg, (short)0, accG[ct], false, false);
    }
    __syncthreads();
  }
#pragma unroll
  for (int ct = 0; ct < 4; ++ct) {
    int ch   = c0 + ct * 16 + ln;
    float ba = bout[ch];
    float bg = bout[DMODEL + ch];
#pragma unroll
    for (int r = 0; r < 8; ++r) {
      size_t rowi = (size_t)(row0 + m0 + r + 8 * hi);
      float a = accA[ct][r] + ba;
      float g = accG[ct][r] + bg;
      float glu = a * (1.0f / (1.0f + expf(-g)));
      rbuf[rowi * DMODEL + ch] = glu + hres[rowi * DMODEL + ch];
    }
  }
}

// ---------------------------------------------------------------------------
// Post-norm layernorm over d=256, one pixel per 256-thread WG.
// ---------------------------------------------------------------------------
__global__ __launch_bounds__(256) void layernorm_kernel(
    const float* __restrict__ rbuf, const float* __restrict__ lnw,
    const float* __restrict__ lnb, float* __restrict__ h,
    _Float16* __restrict__ h16) {
  __shared__ float red[256];
  int pix = blockIdx.x;
  int c   = threadIdx.x;
  float v = rbuf[(size_t)pix * DMODEL + c];
  red[c] = v;
  __syncthreads();
  for (int s = 128; s > 0; s >>= 1) {
    if (c < s) red[c] += red[c + s];
    __syncthreads();
  }
  float mu = red[0] * (1.0f / DMODEL);
  __syncthreads();
  float d = v - mu;
  red[c] = d * d;
  __syncthreads();
  for (int s = 128; s > 0; s >>= 1) {
    if (c < s) red[c] += red[c + s];
    __syncthreads();
  }
  float var = red[0] * (1.0f / DMODEL);
  float y = d * rsqrtf(var + 1e-5f) * lnw[c] + lnb[c];
  h[(size_t)pix * DMODEL + c]   = y;
  h16[(size_t)pix * DMODEL + c] = (_Float16)y;
}

// ---------------------------------------------------------------------------
// Decoder: out[pix] = h[pix, :] . W_dec + b_dec
// ---------------------------------------------------------------------------
__global__ __launch_bounds__(256) void decoder_kernel(
    const float* __restrict__ h, const float* __restrict__ Wdec,
    const float* __restrict__ bdec, float* __restrict__ out) {
  int pix = blockIdx.x * 256 + threadIdx.x;
  float s = bdec[0];
  for (int c = 0; c < DMODEL; ++c) s += h[(size_t)pix * DMODEL + c] * Wdec[c];
  out[pix] = s;
}

// ---------------------------------------------------------------------------
extern "C" void kernel_launch(void* const* d_in, const int* in_sizes, int n_in,
                              void* d_out, int out_size, void* d_ws,
                              size_t ws_size, hipStream_t stream) {
  (void)in_sizes; (void)n_in; (void)out_size; (void)ws_size;
  const float* x       = (const float*)d_in[0];
  const float* grid    = (const float*)d_in[1];
  const float* W_enc   = (const float*)d_in[2];
  const float* b_enc   = (const float*)d_in[3];
  const float* log_dt  = (const float*)d_in[4];
  const float* logA_re = (const float*)d_in[5];
  const float* A_im    = (const float*)d_in[6];
  const float* C_re    = (const float*)d_in[7];
  const float* C_im    = (const float*)d_in[8];
  const float* Dskip   = (const float*)d_in[9];
  const float* W_out   = (const float*)d_in[10];
  const float* b_out   = (const float*)d_in[11];
  const float* ln_w    = (const float*)d_in[12];
  const float* ln_b    = (const float*)d_in[13];
  const float* W_dec   = (const float*)d_in[14];
  const float* b_dec   = (const float*)d_in[15];

  char* ws = (char*)d_ws;
  float*    h    = (float*)(ws);                          // 64 MB f32 residual
  _Float16* h16  = (_Float16*)(ws + ((size_t)64 << 20));  // 32 MB f16 copy
  _Float16* u16  = (_Float16*)(ws + ((size_t)96 << 20));  // 32 MB conv input planes
  _Float16* y1t  = (_Float16*)(ws + ((size_t)128 << 20)); // 32 MB H-pass output
  _Float16* g16  = (_Float16*)(ws + ((size_t)160 << 20)); // 32 MB GELU output
  float*    rbuf = (float*)(ws + ((size_t)96 << 20));     // 64 MB, aliases u16+y1t (dead)
  float*    kf   = (float*)(ws + ((size_t)192 << 20));    // 256 KB SSM kernels

  encoder_kernel<<<NPIX * DMODEL / 256, 256, 0, stream>>>(x, grid, W_enc, b_enc, h, h16);

  for (int i = 0; i < NLAYERS; ++i) {
    size_t po = (size_t)i * 2 * DMODEL * NSTATE;
    ssm_kernel<<<2 * DMODEL * HDIM / 256, 256, 0, stream>>>(
        log_dt + (size_t)i * 2 * DMODEL, logA_re + po, A_im + po, C_re + po,
        C_im + po, kf);

    dim3 tgrid(HW / 32, DMODEL / 32, BATCH);
    transpose_hd_kernel<<<tgrid, dim3(32, 8), 0, stream>>>(h16, u16);

    // H-axis Toeplitz conv: y1t = (T_h @ U)^T
    conv_pass_kernel<<<NPLANES, 256, 0, stream>>>(
        u16, kf, y1t, (const _Float16*)nullptr, (const float*)nullptr,
        (_Float16*)nullptr, 0);
    // W-axis Toeplitz conv + Dskip + GELU -> g16 (pixel-major)
    conv_pass_kernel<<<NPLANES, 256, 0, stream>>>(
        y1t, kf + DMODEL * HDIM, y1t, u16, Dskip + (size_t)i * DMODEL, g16, 1);

    glu_gemm_kernel<<<dim3(NPIX / 128, DMODEL / 64), 256, 0, stream>>>(
        g16, W_out + (size_t)i * DMODEL * 2 * DMODEL,
        b_out + (size_t)i * 2 * DMODEL, h, rbuf);

    layernorm_kernel<<<NPIX, 256, 0, stream>>>(
        rbuf, ln_w + (size_t)i * DMODEL, ln_b + (size_t)i * DMODEL, h, h16);
  }

  decoder_kernel<<<NPIX / 256, 256, 0, stream>>>(h, W_dec, b_dec, (float*)d_out);
}
